// ComplexGRUCell_43413529428634
// MI455X (gfx1250) — compile-verified, matrix operand order checked by reading, not run on hardware
//
#include <hip/hip_runtime.h>
#include <stdint.h>

// ---------------------------------------------------------------------------
// ComplexGRUCell on MI455X (gfx1250), bf16 WMMA formulation.
//   B = 65536 rows, D = 256.
//   Stage 1 GEMM: [B,1024] x [1024,1024] -> gates z_r,z_i,r_r,r_i (sigmoid),
//                 fused epilogue computes rh = r (*) h (complex elementwise).
//   Stage 2 GEMM: [B,1024] x [1024,512] -> n_r,n_i (tanh), fused epilogue
//                 computes new_h = (1-z)*h + z*n and writes d_out.
// Wave tile: M=64 (4x16), N=16, {4 gates | 2 components} -> B fragments are
// amortized over 4 M-tiles: 16 vmem / 16 wmma per K-step in GEMM1.
// ---------------------------------------------------------------------------

#define NB    65536
#define NDIM  256

typedef __attribute__((ext_vector_type(16))) __bf16  v16bf;
typedef __attribute__((ext_vector_type(8)))  float   v8f;
typedef __attribute__((ext_vector_type(4)))  unsigned int u32x4;
typedef __attribute__((ext_vector_type(2)))  unsigned int u32x2;
typedef __attribute__((ext_vector_type(4)))  float   f32x4;

union FragBF {
    u32x4 u[2];
    v16bf v;
};

__device__ __forceinline__ unsigned short f2bf(float f) {
    union { float f; unsigned u; } x; x.f = f;
    unsigned r = x.u + 0x7FFFu + ((x.u >> 16) & 1u);   // round-to-nearest-even
    return (unsigned short)(r >> 16);
}
__device__ __forceinline__ float bf2f(unsigned short s) {
    union { unsigned u; float f; } x; x.u = ((unsigned)s) << 16;
    return x.f;
}
__device__ __forceinline__ float sigmoidf(float x) {
    return 1.0f / (1.0f + __expf(-x));
}

// --------------------------- weight packing --------------------------------
// W1t: [1024 (n = 4 gates x 256)] x [1024 (k = h_r|h_i|x_r|x_i)] bf16, N-major.
__global__ void pack_w1(const float* __restrict__ h_zr, const float* __restrict__ h_zi,
                        const float* __restrict__ x_zr, const float* __restrict__ x_zi,
                        const float* __restrict__ h_rr, const float* __restrict__ h_ri,
                        const float* __restrict__ x_rr, const float* __restrict__ x_ri,
                        unsigned short* __restrict__ W1t) {
    int idx = blockIdx.x * blockDim.x + threadIdx.x;   // 1024*1024 total
    int n = idx >> 10, k = idx & 1023;
    int g = n >> 8,  j = n & 255;
    int s = k >> 8,  kk = k & 255;
    const float* tbl[4][4] = {
        { h_zr, h_zi, x_zr, x_zi },   // z_r :  +h_zr  -h_zi  +x_zr  -x_zi
        { h_zi, h_zr, x_zi, x_zr },   // z_i :  +h_zi  +h_zr  +x_zi  +x_zr
        { h_rr, h_ri, x_rr, x_ri },   // r_r :  +h_rr  -h_ri  +x_rr  -x_ri
        { h_ri, h_rr, x_ri, x_rr } }; // r_i :  all +
    float sgn = (((g & 1) == 0) && (s & 1)) ? -1.0f : 1.0f;
    W1t[idx] = f2bf(sgn * tbl[g][s][kk * NDIM + j]);
}

// W2t: [512 (n = n_r|n_i)] x [1024 (k = rh_r|rh_i|x_r|x_i)] bf16, N-major.
__global__ void pack_w2(const float* __restrict__ h_nr, const float* __restrict__ h_ni,
                        const float* __restrict__ x_nr, const float* __restrict__ x_ni,
                        unsigned short* __restrict__ W2t) {
    int idx = blockIdx.x * blockDim.x + threadIdx.x;   // 512*1024 total
    int n = idx >> 10, k = idx & 1023;
    int c = n >> 8,  j = n & 255;
    int s = k >> 8,  kk = k & 255;
    const float* tbl[2][4] = {
        { h_nr, h_ni, x_nr, x_ni },   // n_r :  +h_nr  -h_ni  +x_nr  -x_ni
        { h_ni, h_nr, x_ni, x_nr } }; // n_i :  all +
    float sgn = ((c == 0) && (s & 1)) ? -1.0f : 1.0f;
    W2t[idx] = f2bf(sgn * tbl[c][s][kk * NDIM + j]);
}

// A1 = [h_r | h_i | x_r | x_i] as bf16, row-major [B,1024].
__global__ void cvt_inputs(const float* __restrict__ h_r, const float* __restrict__ h_i,
                           const float* __restrict__ x_r, const float* __restrict__ x_i,
                           unsigned short* __restrict__ A1) {
    int idx = (blockIdx.x * blockDim.x + threadIdx.x) * 4;  // element in [B,1024]
    int row = idx >> 10;
    int c   = idx & 1023;
    int s = c >> 8, cc = c & 255;
    const float* srcs[4] = { h_r, h_i, x_r, x_i };
    f32x4 v = *reinterpret_cast<const f32x4*>(srcs[s] + row * NDIM + cc);
    u32x2 o;
    o.x = (unsigned)f2bf(v.x) | ((unsigned)f2bf(v.y) << 16);
    o.y = (unsigned)f2bf(v.z) | ((unsigned)f2bf(v.w) << 16);
    *reinterpret_cast<u32x2*>(A1 + idx) = o;
}

// ------------------------------- GEMM 1 ------------------------------------
// grid = (8 Nblocks, 256 rowblocks), 256 threads = 8 waves (4 Msub x 2 Nsub).
// Wave: M=64 (4x16 tiles), N=16, x4 gates -> 16 accum tiles, 512 WMMAs.
__global__ void __launch_bounds__(256)
gemm1_gates(const unsigned short* __restrict__ A1,   // [B,1024] bf16
            const unsigned short* __restrict__ W1t,  // [1024,1024] bf16
            const float* __restrict__ zb_r, const float* __restrict__ zb_i,
            const float* __restrict__ rb_r, const float* __restrict__ rb_i,
            const float* __restrict__ h_r,  const float* __restrict__ h_i,
            unsigned short* __restrict__ zbuf,       // [B,512]  z_r|z_i
            unsigned short* __restrict__ A2rh) {     // [B,512]  rh_r|rh_i
    const int lane = threadIdx.x & 31;
    const int wave = threadIdx.x >> 5;
    const int half = lane >> 4;
    const int l16  = lane & 15;
    const int wm = wave & 3;
    const int wn = wave >> 2;
    const int rowBase = blockIdx.y * 256 + wm * 64;
    const int nBase   = blockIdx.x * 32  + wn * 16;

    v8f zero = {};
    v8f acc[4][4];
#pragma unroll
    for (int t = 0; t < 4; ++t)
#pragma unroll
        for (int g = 0; g < 4; ++g) acc[t][g] = zero;

    const unsigned short* aptr[4];
#pragma unroll
    for (int t = 0; t < 4; ++t)
        aptr[t] = A1 + (rowBase + t * 16 + l16) * 1024 + half * 8;
    const unsigned short* bptr[4];
#pragma unroll
    for (int g = 0; g < 4; ++g)
        bptr[g] = W1t + (g * 256 + nBase + l16) * 1024 + half * 8;

    for (int kBase = 0; kBase < 1024; kBase += 32) {
        if (kBase + 256 < 1024) {
            __builtin_prefetch(aptr[0] + kBase + 256, 0, 1);  // global_prefetch
            __builtin_prefetch(aptr[2] + kBase + 256, 0, 1);
        }
        FragBF a[4], b[4];
#pragma unroll
        for (int t = 0; t < 4; ++t) {
            const u32x4* p = reinterpret_cast<const u32x4*>(aptr[t] + kBase);
            a[t].u[0] = p[0]; a[t].u[1] = p[2];
        }
#pragma unroll
        for (int g = 0; g < 4; ++g) {
            const u32x4* pb = reinterpret_cast<const u32x4*>(bptr[g] + kBase);
            b[g].u[0] = pb[0]; b[g].u[1] = pb[2];
        }
#pragma unroll
        for (int g = 0; g < 4; ++g)
#pragma unroll
            for (int t = 0; t < 4; ++t)
                acc[t][g] = __builtin_amdgcn_wmma_f32_16x16x32_bf16(
                    false, a[t].v, false, b[g].v, (short)0, acc[t][g], false, false);
    }

    // fused epilogue: sigmoid gates, store z, compute rh = r (*) h
    const int j = nBase + l16;
    const float zbr = zb_r[j], zbi = zb_i[j], rbr = rb_r[j], rbi = rb_i[j];
#pragma unroll
    for (int t = 0; t < 4; ++t) {
#pragma unroll
        for (int v = 0; v < 8; ++v) {
            int row = rowBase + t * 16 + half * 8 + v;
            float zr = sigmoidf(acc[t][0][v] + zbr);
            float zi = sigmoidf(acc[t][1][v] + zbi);
            float rr = sigmoidf(acc[t][2][v] + rbr);
            float ri = sigmoidf(acc[t][3][v] + rbi);
            float hr = h_r[row * NDIM + j];
            float hi = h_i[row * NDIM + j];
            zbuf[row * 512 + j]        = f2bf(zr);
            zbuf[row * 512 + 256 + j]  = f2bf(zi);
            A2rh[row * 512 + j]        = f2bf(rr * hr - ri * hi);
            A2rh[row * 512 + 256 + j]  = f2bf(rr * hi + ri * hr);
        }
    }
}

// ------------------------------- GEMM 2 ------------------------------------
// A2cat = [rh_r | rh_i] from A2rh (stride 512) and [x_r | x_i] from A1 cols
// 512..1023 (stride 1024). Wave computes n_r and n_i tiles for the same j.
__global__ void __launch_bounds__(256)
gemm2_out(const unsigned short* __restrict__ A2rh,  // [B,512] bf16
          const unsigned short* __restrict__ A1,    // [B,1024] bf16
          const unsigned short* __restrict__ W2t,   // [512,1024] bf16
          const float* __restrict__ nb_r, const float* __restrict__ nb_i,
          const unsigned short* __restrict__ zbuf,  // [B,512] bf16
          const float* __restrict__ h_r,  const float* __restrict__ h_i,
          float* __restrict__ out_r, float* __restrict__ out_i) {
    const int lane = threadIdx.x & 31;
    const int wave = threadIdx.x >> 5;
    const int half = lane >> 4;
    const int l16  = lane & 15;
    const int wm = wave & 3;
    const int wn = wave >> 2;
    const int rowBase = blockIdx.y * 256 + wm * 64;
    const int nBase   = blockIdx.x * 32  + wn * 16;

    v8f zero = {};
    v8f acc[4][2];
#pragma unroll
    for (int t = 0; t < 4; ++t) { acc[t][0] = zero; acc[t][1] = zero; }

    const unsigned short* bptr0 = W2t + (nBase + l16) * 1024 + half * 8;
    const unsigned short* bptr1 = W2t + (256 + nBase + l16) * 1024 + half * 8;

    for (int kBase = 0; kBase < 1024; kBase += 32) {
        FragBF a[4], b0, b1;
        if (kBase < 512) {  // rh part, stride 512
#pragma unroll
            for (int t = 0; t < 4; ++t) {
                const u32x4* p = reinterpret_cast<const u32x4*>(
                    A2rh + (rowBase + t * 16 + l16) * 512 + kBase + half * 8);
                a[t].u[0] = p[0]; a[t].u[1] = p[2];
            }
        } else {            // x part lives in A1 cols 512..1023, stride 1024
#pragma unroll
            for (int t = 0; t < 4; ++t) {
                const u32x4* p = reinterpret_cast<const u32x4*>(
                    A1 + (rowBase + t * 16 + l16) * 1024 + kBase + half * 8);
                a[t].u[0] = p[0]; a[t].u[1] = p[2];
            }
        }
        const u32x4* pb0 = reinterpret_cast<const u32x4*>(bptr0 + kBase);
        const u32x4* pb1 = reinterpret_cast<const u32x4*>(bptr1 + kBase);
        b0.u[0] = pb0[0]; b0.u[1] = pb0[2];
        b1.u[0] = pb1[0]; b1.u[1] = pb1[2];

#pragma unroll
        for (int t = 0; t < 4; ++t) {
            acc[t][0] = __builtin_amdgcn_wmma_f32_16x16x32_bf16(
                false, a[t].v, false, b0.v, (short)0, acc[t][0], false, false);
            acc[t][1] = __builtin_amdgcn_wmma_f32_16x16x32_bf16(
                false, a[t].v, false, b1.v, (short)0, acc[t][1], false, false);
        }
    }

    // fused epilogue: tanh, then new_h = (1-z)*h + z*n (complex)
    const int j = nBase + l16;
    const float nbr = nb_r[j], nbi = nb_i[j];
#pragma unroll
    for (int t = 0; t < 4; ++t) {
#pragma unroll
        for (int v = 0; v < 8; ++v) {
            int row = rowBase + t * 16 + half * 8 + v;
            float nr = tanhf(acc[t][0][v] + nbr);
            float ni = tanhf(acc[t][1][v] + nbi);
            float zr = bf2f(zbuf[row * 512 + j]);
            float zi = bf2f(zbuf[row * 512 + 256 + j]);
            float hr = h_r[row * NDIM + j];
            float hi = h_i[row * NDIM + j];
            float t1r = (1.0f - zr) * hr + zi * hi;
            float t1i = (1.0f - zr) * hi - zi * hr;
            float t2r = zr * nr - zi * ni;
            float t2i = zr * ni + zi * nr;
            out_r[row * NDIM + j] = t1r + t2r;
            out_i[row * NDIM + j] = t1i + t2i;
        }
    }
}

// ----------------------------- launcher ------------------------------------
extern "C" void kernel_launch(void* const* d_in, const int* in_sizes, int n_in,
                              void* d_out, int out_size, void* d_ws, size_t ws_size,
                              hipStream_t stream) {
    const float* h_r = (const float*)d_in[0];
    const float* h_i = (const float*)d_in[1];
    const float* x_r = (const float*)d_in[2];
    const float* x_i = (const float*)d_in[3];
    // params in dict order: h_zr h_zi x_zr x_zi h_rr h_ri x_rr x_ri h_nr h_ni x_nr x_ni
    const float* h_zr = (const float*)d_in[4];
    const float* h_zi = (const float*)d_in[5];
    const float* x_zr = (const float*)d_in[6];
    const float* x_zi = (const float*)d_in[7];
    const float* h_rr = (const float*)d_in[8];
    const float* h_ri = (const float*)d_in[9];
    const float* x_rr = (const float*)d_in[10];
    const float* x_ri = (const float*)d_in[11];
    const float* h_nr = (const float*)d_in[12];
    const float* h_ni = (const float*)d_in[13];
    const float* x_nr = (const float*)d_in[14];
    const float* x_ni = (const float*)d_in[15];
    const float* zb_r = (const float*)d_in[16];
    const float* zb_i = (const float*)d_in[17];
    const float* rb_r = (const float*)d_in[18];
    const float* rb_i = (const float*)d_in[19];
    const float* nb_r = (const float*)d_in[20];
    const float* nb_i = (const float*)d_in[21];

    char* ws = (char*)d_ws;
    unsigned short* W1t  = (unsigned short*)(ws);                         //   2 MiB
    unsigned short* W2t  = (unsigned short*)(ws + (size_t)2  * 1048576);  //   1 MiB
    unsigned short* A1   = (unsigned short*)(ws + (size_t)3  * 1048576);  // 128 MiB
    unsigned short* A2rh = (unsigned short*)(ws + (size_t)131 * 1048576); //  64 MiB
    unsigned short* zbuf = (unsigned short*)(ws + (size_t)195 * 1048576); //  64 MiB

    float* out_r = (float*)d_out;
    float* out_i = out_r + (size_t)NB * NDIM;

    pack_w1<<<4096, 256, 0, stream>>>(h_zr, h_zi, x_zr, x_zi,
                                      h_rr, h_ri, x_rr, x_ri, W1t);
    pack_w2<<<2048, 256, 0, stream>>>(h_nr, h_ni, x_nr, x_ni, W2t);
    cvt_inputs<<<65536, 256, 0, stream>>>(h_r, h_i, x_r, x_i, A1);

    gemm1_gates<<<dim3(8, 256), 256, 0, stream>>>(A1, W1t,
                                                  zb_r, zb_i, rb_r, rb_i,
                                                  h_r, h_i, zbuf, A2rh);
    gemm2_out<<<dim3(8, 256), 256, 0, stream>>>(A2rh, A1, W2t,
                                                nb_r, nb_i, zbuf,
                                                h_r, h_i, out_r, out_i);
}
